// HydraBlock_9191230014092
// MI455X (gfx1250) — compile-verified, matrix operand order checked by reading, not run on hardware
//
#include <hip/hip_runtime.h>
#include <hip/hip_bf16.h>

// ---------------------------------------------------------------------------
// Types for gfx1250 WMMA (wave32): v_wmma_f32_16x16x32_bf16
// ---------------------------------------------------------------------------
typedef __bf16 bf16;
typedef __attribute__((ext_vector_type(16))) __bf16 v16bf;
typedef __attribute__((ext_vector_type(8)))  __bf16 v8bf;
typedef __attribute__((ext_vector_type(8)))  float  v8f;

// Pointer types for the async global->LDS builtin:
//   param0: int4 in addrspace(1) (global), param1: int4 in addrspace(3) (LDS)
typedef int v4i __attribute__((vector_size(16)));
typedef __attribute__((address_space(1))) v4i* gv4i_p;
typedef __attribute__((address_space(3))) v4i* lv4i_p;

#define DMODEL 1024
#define NTOK   8192   // B*T = 4*2048
#define NEXP   8

// ---- CDNA5 async global->LDS staging (ASYNCcnt path), guarded ------------
#if defined(__has_builtin)
#  if __has_builtin(__builtin_amdgcn_global_load_async_to_lds_b128)
#    define HAVE_ASYNC_LDS 1
#  endif
#endif
#ifndef HAVE_ASYNC_LDS
#  define HAVE_ASYNC_LDS 0
#endif

#if HAVE_ASYNC_LDS
#  define STAGE_B128(g, l)                                                    \
      __builtin_amdgcn_global_load_async_to_lds_b128(                         \
          (gv4i_p)(g), (lv4i_p)(l), 0, 0)
#  if __has_builtin(__builtin_amdgcn_s_wait_asynccnt)
#    define WAIT_ASYNC() __builtin_amdgcn_s_wait_asynccnt(0)
#  else
#    define WAIT_ASYNC() asm volatile("s_wait_asynccnt 0" ::: "memory")
#  endif
#else
#  define STAGE_B128(g, l) (*(v8bf*)(l) = *(const v8bf*)(g))
#  define WAIT_ASYNC()
#endif

__device__ __forceinline__ bf16 f2bf(float f) {
    unsigned u = __builtin_bit_cast(unsigned, f);
    unsigned r = u + 0x7fffu + ((u >> 16) & 1u);   // round-to-nearest-even
    unsigned short h = (unsigned short)(r >> 16);
    return __builtin_bit_cast(bf16, h);
}
__device__ __forceinline__ float bf2f(bf16 b) {
    unsigned u = ((unsigned)__builtin_bit_cast(unsigned short, b)) << 16;
    return __builtin_bit_cast(float, u);
}
__device__ __forceinline__ v8f vzero8() { v8f z = {0,0,0,0,0,0,0,0}; return z; }

union AB16 { v16bf v; v8bf h[2]; };

__device__ __forceinline__ v8f wmma_bf16(v16bf a, v16bf b, v8f c) {
    return __builtin_amdgcn_wmma_f32_16x16x32_bf16(false, a, false, b,
                                                   (short)0, c, false, false);
}

// ---------------------------------------------------------------------------
// Weight transpose + fp32->bf16 convert:  dst[b][c][r] = src[b][r][c]
// ---------------------------------------------------------------------------
__global__ void k_transpose_cvt(const float* __restrict__ src, bf16* __restrict__ dst,
                                int R, int C) {
    long bo = (long)blockIdx.z * R * C;
    int idx = blockIdx.x * blockDim.x + threadIdx.x;
    if (idx < R * C) {
        int r = idx / C, c = idx % C;
        dst[bo + (long)c * R + r] = f2bf(src[bo + idx]);
    }
}

// ---------------------------------------------------------------------------
// RMSNorm (D=1024) -> bf16
// ---------------------------------------------------------------------------
__global__ __launch_bounds__(256)
void k_rmsnorm_bf16(const float* __restrict__ x, const float* __restrict__ gamma,
                    bf16* __restrict__ out) {
    __shared__ float red[256];
    long t = blockIdx.x;
    const float* row = x + t * DMODEL;
    float ss = 0.f;
#pragma unroll
    for (int i = 0; i < 4; ++i) { float v = row[threadIdx.x + i*256]; ss += v*v; }
    red[threadIdx.x] = ss;
    __syncthreads();
    for (int s = 128; s > 0; s >>= 1) {
        if (threadIdx.x < s) red[threadIdx.x] += red[threadIdx.x + s];
        __syncthreads();
    }
    float rinv = rsqrtf(red[0] * (1.0f / DMODEL) + 1e-6f);
    bf16* orow = out + t * DMODEL;
#pragma unroll
    for (int i = 0; i < 4; ++i) {
        int d = threadIdx.x + i*256;
        orow[d] = f2bf(gamma[d] * row[d] * rinv);
    }
}

// ---------------------------------------------------------------------------
// WMMA GEMM with LDS-staged, double-buffered B tiles (async global->LDS).
//   C[M,N] = A[M,K](bf16) * BT[N,K](bf16)^T (+bias) (+residual)
// Block = 256 threads = 8 waves stacked along M -> block tile M128 x N64.
// B tile (64 rows x 32 K, 4 KB) staged once per K-step, shared by 8 waves.
// Grid = (N/64, M/128).
// ---------------------------------------------------------------------------
template<bool OUT_BF16, bool HAS_BIAS, bool HAS_RES>
__global__ __launch_bounds__(256)
void k_gemm_bf16(const bf16* __restrict__ A, const bf16* __restrict__ BT,
                 const float* __restrict__ bias, const float* __restrict__ res,
                 void* __restrict__ outp, int M, int N, int K) {
    __shared__ bf16 bstage[2][64 * 32];
    const int tid    = threadIdx.x;
    const int lane   = tid & 31;
    const int wave   = tid >> 5;
    const int lanelo = lane & 15;
    const int kb     = (lane < 16) ? 0 : 8;
    const int rowM   = blockIdx.y * 128 + wave * 16;
    const int colN   = blockIdx.x * 64;
    const int srow   = tid >> 2;          // staged row 0..63
    const int schunk = (tid & 3) * 8;     // 16B chunk within 32-wide K slab

    v8f acc[4];
#pragma unroll
    for (int j = 0; j < 4; ++j) acc[j] = vzero8();

    const bf16* arow = A + (long)(rowM + lanelo) * K;
    const bf16* bsrc = BT + (long)(colN + srow) * K + schunk;
    bf16* sdst[2] = { &bstage[0][srow * 32 + schunk], &bstage[1][srow * 32 + schunk] };

    STAGE_B128(bsrc, sdst[0]);                       // prologue: K-step 0
    const int nsteps = K >> 5;
    for (int s = 0; s < nsteps; ++s) {
        WAIT_ASYNC();
        __syncthreads();                              // buf[s&1] ready; prior reads done
        if (s + 1 < nsteps)
            STAGE_B128(bsrc + ((long)(s + 1) << 5), sdst[(s + 1) & 1]);
        const int k0 = s << 5;
        if (s + 4 < nsteps) __builtin_prefetch(arow + ((long)(s + 4) << 5), 0, 0);
        AB16 a;
        a.h[0] = *(const v8bf*)(arow + k0 + kb);
        a.h[1] = *(const v8bf*)(arow + k0 + kb + 16);
        const bf16* buf = bstage[s & 1];
#pragma unroll
        for (int j = 0; j < 4; ++j) {
            AB16 b;
            b.h[0] = *(const v8bf*)(buf + (j*16 + lanelo) * 32 + kb);
            b.h[1] = *(const v8bf*)(buf + (j*16 + lanelo) * 32 + kb + 16);
            acc[j] = wmma_bf16(a.v, b.v, acc[j]);
        }
    }
    // C layout: lane<16 -> rows rowM+0..7, lane>=16 -> rows rowM+8..15
    const int mB = rowM + ((lane < 16) ? 0 : 8);
#pragma unroll
    for (int j = 0; j < 4; ++j) {
        int n = colN + j*16 + lanelo;
        float bval = HAS_BIAS ? bias[n] : 0.f;
#pragma unroll
        for (int v = 0; v < 8; ++v) {
            long m = mB + v;
            float val = acc[j][v] + bval;
            if (HAS_RES) val += res[m * N + n];
            if (OUT_BF16) ((bf16*)outp)[m * N + n] = f2bf(val);
            else          ((float*)outp)[m * N + n] = val;
        }
    }
}

// ---------------------------------------------------------------------------
// Per-token MLA attention. One wave per token (4 waves / 128-thread block).
//   q_t[h,d]  = q[t*1024 + h*64 + d]            (16x64)
//   k_t[g,d]  = kv[t*2048 + g*128 + d]          (16x64, interleaved split)
//   v_t[g,d]  = kv[t*2048 + g*128 + 64 + d]
//   att = softmax_g(q_t k_t^T / 8); out = att v_t   (16x64)
// ---------------------------------------------------------------------------
__global__ __launch_bounds__(128)
void k_attention(const bf16* __restrict__ q, const bf16* __restrict__ kv,
                 bf16* __restrict__ outp) {
    __shared__ unsigned short att_s[4][256];   // per-wave 16x16 bf16 tile
    const int lane   = threadIdx.x & 31;
    const int wave   = threadIdx.x >> 5;
    const int lanelo = lane & 15;
    const int kb     = (lane < 16) ? 0 : 8;
    const long t     = (long)blockIdx.x * 4 + wave;

    const bf16* qrow = q  + t * 1024 + (long)lanelo * 64;
    const bf16* krow = kv + t * 2048 + (long)lanelo * 128;

    v8f att = vzero8();
#pragma unroll
    for (int k0 = 0; k0 < 64; k0 += 32) {
        AB16 a, b;
        a.h[0] = *(const v8bf*)(qrow + k0 + kb);
        a.h[1] = *(const v8bf*)(qrow + k0 + kb + 16);
        b.h[0] = *(const v8bf*)(krow + k0 + kb);
        b.h[1] = *(const v8bf*)(krow + k0 + kb + 16);
        att = wmma_bf16(a.v, b.v, att);
    }
    const int mrow = (lane < 16) ? 0 : 8;
#pragma unroll
    for (int v = 0; v < 8; ++v) {
        float val = att[v] * 0.125f;   // 1/sqrt(64)
        float mx = val;
#pragma unroll
        for (int off = 8; off; off >>= 1) mx = fmaxf(mx, __shfl_xor(mx, off, 16));
        float e = __expf(val - mx);
        float s = e;
#pragma unroll
        for (int off = 8; off; off >>= 1) s += __shfl_xor(s, off, 16);
        att_s[wave][(mrow + v) * 16 + lanelo] =
            __builtin_bit_cast(unsigned short, f2bf(e / s));
    }
    __syncthreads();

    const bf16 BZ = __builtin_bit_cast(bf16, (unsigned short)0);
    AB16 a2;
    a2.h[0] = *(const v8bf*)&att_s[wave][lanelo * 16 + kb];
#pragma unroll
    for (int i = 0; i < 8; ++i) a2.h[1][i] = BZ;

#pragma unroll
    for (int j = 0; j < 4; ++j) {
        int d = j * 16 + lanelo;
        v16bf bv;
#pragma unroll
        for (int i = 0; i < 8; ++i)
            bv[i] = kv[t * 2048 + (long)(kb + i) * 128 + 64 + d];
#pragma unroll
        for (int i = 8; i < 16; ++i) bv[i] = BZ;
        v8f o = wmma_bf16(a2.v, bv, vzero8());
#pragma unroll
        for (int v = 0; v < 8; ++v)
            outp[t * 1024 + (long)(mrow + v) * 64 + d] = f2bf(o[v]);
    }
}

// ---------------------------------------------------------------------------
// Gate: softmax(h2 @ Wg + bg) over E=8. One block per token.
// ---------------------------------------------------------------------------
__global__ __launch_bounds__(256)
void k_gate(const bf16* __restrict__ h2, const float* __restrict__ Wg,
            const float* __restrict__ bg, float* __restrict__ gate) {
    __shared__ float part[8][8];   // [wave][expert]
    long t = blockIdx.x;
    float p[8] = {0,0,0,0,0,0,0,0};
#pragma unroll
    for (int i = 0; i < 4; ++i) {
        int d = threadIdx.x + i * 256;
        float xv = bf2f(h2[t * 1024 + d]);
#pragma unroll
        for (int e = 0; e < 8; ++e) p[e] += xv * Wg[(long)d * 8 + e];
    }
#pragma unroll
    for (int e = 0; e < 8; ++e)
        for (int off = 16; off; off >>= 1) p[e] += __shfl_xor(p[e], off, 32);
    int wave = threadIdx.x >> 5, lane = threadIdx.x & 31;
    if (lane == 0)
        for (int e = 0; e < 8; ++e) part[wave][e] = p[e];
    __syncthreads();
    if (threadIdx.x == 0) {
        float lg[8], mx = -3.4e38f;
        for (int e = 0; e < 8; ++e) {
            float s = bg[e];
            for (int w = 0; w < 8; ++w) s += part[w][e];
            lg[e] = s; mx = fmaxf(mx, s);
        }
        float den = 0.f;
        for (int e = 0; e < 8; ++e) { lg[e] = __expf(lg[e] - mx); den += lg[e]; }
        for (int e = 0; e < 8; ++e) gate[t * 8 + e] = lg[e] / den;
    }
}

// ---------------------------------------------------------------------------
// Fused dense MoE with LDS-staged expert weights:
//   out = x1 + sum_e gate[:,e] * (h2 @ We[e])
// Flattened (expert, K-step) pipeline over a double-buffered 4 KB B tile.
// 1024 WMMAs per wave; gate scaling applied in-register per expert.
// ---------------------------------------------------------------------------
__global__ __launch_bounds__(256)
void k_moe(const bf16* __restrict__ h2, const bf16* __restrict__ WeT,
           const float* __restrict__ gate, const float* __restrict__ x1,
           float* __restrict__ outp) {
    __shared__ bf16 bstage[2][64 * 32];
    const int tid    = threadIdx.x;
    const int lane   = tid & 31;
    const int wave   = tid >> 5;
    const int lanelo = lane & 15;
    const int kb     = (lane < 16) ? 0 : 8;
    const int rowM   = blockIdx.y * 128 + wave * 16;
    const int colN   = blockIdx.x * 64;
    const int srow   = tid >> 2;
    const int schunk = (tid & 3) * 8;
    const int mB     = rowM + ((lane < 16) ? 0 : 8);

    v8f accF[4];
#pragma unroll
    for (int j = 0; j < 4; ++j) accF[j] = vzero8();

    const bf16* arow  = h2 + (long)(rowM + lanelo) * DMODEL;
    const bf16* bbase = WeT + (long)(colN + srow) * DMODEL + schunk; // expert 0
    bf16* sdst[2] = { &bstage[0][srow * 32 + schunk], &bstage[1][srow * 32 + schunk] };

    STAGE_B128(bbase, sdst[0]);   // (e=0, k=0)
    int s = 0;
    const int NSTEPS = NEXP * (DMODEL >> 5);   // 256
    for (int e = 0; e < NEXP; ++e) {
        v8f accE[4];
#pragma unroll
        for (int j = 0; j < 4; ++j) accE[j] = vzero8();
        for (int kk = 0; kk < (DMODEL >> 5); ++kk, ++s) {
            WAIT_ASYNC();
            __syncthreads();
            if (s + 1 < NSTEPS) {
                int e2 = (s + 1) >> 5;
                int k2 = ((s + 1) & 31) << 5;
                STAGE_B128(bbase + (long)e2 * DMODEL * DMODEL + k2, sdst[(s + 1) & 1]);
            }
            const int k0 = kk << 5;
            if (kk + 4 < (DMODEL >> 5))
                __builtin_prefetch(arow + ((long)(kk + 4) << 5), 0, 0);
            AB16 a;
            a.h[0] = *(const v8bf*)(arow + k0 + kb);
            a.h[1] = *(const v8bf*)(arow + k0 + kb + 16);
            const bf16* buf = bstage[s & 1];
#pragma unroll
            for (int j = 0; j < 4; ++j) {
                AB16 b;
                b.h[0] = *(const v8bf*)(buf + (j*16 + lanelo) * 32 + kb);
                b.h[1] = *(const v8bf*)(buf + (j*16 + lanelo) * 32 + kb + 16);
                accE[j] = wmma_bf16(a.v, b.v, accE[j]);
            }
        }
#pragma unroll
        for (int v = 0; v < 8; ++v) {
            float g = gate[(long)(mB + v) * NEXP + e];
#pragma unroll
            for (int j = 0; j < 4; ++j) accF[j][v] += g * accE[j][v];
        }
    }
#pragma unroll
    for (int j = 0; j < 4; ++j) {
        int n = colN + j*16 + lanelo;
#pragma unroll
        for (int v = 0; v < 8; ++v) {
            long m = mB + v;
            outp[m * DMODEL + n] = x1[m * DMODEL + n] + accF[j][v];
        }
    }
}

// ---------------------------------------------------------------------------
extern "C" void kernel_launch(void* const* d_in, const int* in_sizes, int n_in,
                              void* d_out, int out_size, void* d_ws, size_t ws_size,
                              hipStream_t stream) {
    (void)in_sizes; (void)n_in; (void)out_size; (void)ws_size;
    const float* x      = (const float*)d_in[0];
    const float* gamma1 = (const float*)d_in[1];
    const float* Wq     = (const float*)d_in[2];
    const float* bq     = (const float*)d_in[3];
    const float* Wdown  = (const float*)d_in[4];
    const float* bdown  = (const float*)d_in[5];
    const float* Wup    = (const float*)d_in[6];
    const float* bup    = (const float*)d_in[7];
    const float* Wo     = (const float*)d_in[8];
    const float* bo     = (const float*)d_in[9];
    const float* gamma2 = (const float*)d_in[10];
    const float* Wg     = (const float*)d_in[11];
    const float* bg     = (const float*)d_in[12];
    const float* We     = (const float*)d_in[13];
    float* out = (float*)d_out;

    char* ws = (char*)d_ws;
    size_t off = 0;
    auto alloc = [&](size_t bytes) -> void* {
        void* p = ws + off;
        off += (bytes + 255) & ~(size_t)255;
        return p;
    };
    bf16*  h_bf    = (bf16*) alloc((size_t)NTOK * 1024 * 2);
    bf16*  q_bf    = (bf16*) alloc((size_t)NTOK * 1024 * 2);
    bf16*  c_bf    = (bf16*) alloc((size_t)NTOK * 64   * 2);
    bf16*  kv_bf   = (bf16*) alloc((size_t)NTOK * 2048 * 2);
    bf16*  att_bf  = (bf16*) alloc((size_t)NTOK * 1024 * 2);
    float* x1      = (float*)alloc((size_t)NTOK * 1024 * 4);
    bf16*  h2_bf   = (bf16*) alloc((size_t)NTOK * 1024 * 2);
    float* gatebuf = (float*)alloc((size_t)NTOK * 8 * 4);
    bf16*  WqT     = (bf16*) alloc((size_t)1024 * 1024 * 2);
    bf16*  WdownT  = (bf16*) alloc((size_t)64   * 1024 * 2);
    bf16*  WupT    = (bf16*) alloc((size_t)2048 * 64   * 2);
    bf16*  WoT     = (bf16*) alloc((size_t)1024 * 1024 * 2);
    bf16*  WeT     = (bf16*) alloc((size_t)8 * 1024 * 1024 * 2);

    // ---- weight convert/transpose (bf16, K-contiguous) ----
    k_transpose_cvt<<<dim3(4096, 1, 1), 256, 0, stream>>>(Wq, WqT, 1024, 1024);
    k_transpose_cvt<<<dim3(256,  1, 1), 256, 0, stream>>>(Wdown, WdownT, 1024, 64);
    k_transpose_cvt<<<dim3(512,  1, 1), 256, 0, stream>>>(Wup, WupT, 64, 2048);
    k_transpose_cvt<<<dim3(4096, 1, 1), 256, 0, stream>>>(Wo, WoT, 1024, 1024);
    k_transpose_cvt<<<dim3(4096, 1, 8), 256, 0, stream>>>(We, WeT, 1024, 1024);

    // ---- MLA attention branch ----
    k_rmsnorm_bf16<<<NTOK, 256, 0, stream>>>(x, gamma1, h_bf);
    k_gemm_bf16<true,  true, false><<<dim3(16, 64), 256, 0, stream>>>(
        h_bf, WqT, bq, nullptr, q_bf, NTOK, 1024, 1024);
    k_gemm_bf16<true,  true, false><<<dim3(1, 64), 256, 0, stream>>>(
        h_bf, WdownT, bdown, nullptr, c_bf, NTOK, 64, 1024);
    k_gemm_bf16<true,  true, false><<<dim3(32, 64), 256, 0, stream>>>(
        c_bf, WupT, bup, nullptr, kv_bf, NTOK, 2048, 64);
    k_attention<<<NTOK / 4, 128, 0, stream>>>(q_bf, kv_bf, att_bf);
    k_gemm_bf16<false, true, true ><<<dim3(16, 64), 256, 0, stream>>>(
        att_bf, WoT, bo, x, x1, NTOK, 1024, 1024);

    // ---- dense gated MoE branch ----
    k_rmsnorm_bf16<<<NTOK, 256, 0, stream>>>(x1, gamma2, h2_bf);
    k_gate<<<NTOK, 256, 0, stream>>>(h2_bf, Wg, bg, gatebuf);
    k_moe<<<dim3(16, 64), 256, 0, stream>>>(h2_bf, WeT, gatebuf, x1, out);
}
